// TrIPModel_36172214567116
// MI455X (gfx1250) — compile-verified, matrix operand order checked by reading, not run on hardware
//
#include <hip/hip_runtime.h>
#include <math.h>

// ---------------------------------------------------------------------------
// TrIP-style equivariant GNN forward for MI455X (gfx1250, wave32, WMMA).
//
// Dominant FLOPs (per-edge generated-weight radial MLPs) restructured as
//   m[e,o] = sum_{k,i} (h[e,k]*f[e,i]) * w2[k,o*16+i] + sum_i b2[o*16+i]*f[e,i]
// i.e. a shared GEMM (16edges x 512) @ (512 x 16): 19 v_wmma_f32_16x16x32_f16
// per 16-edge tile. Weight B-fragments are pre-packed per launch into the
// exact WMMA lane layout so the hot loop issues only b128 loads.
// ---------------------------------------------------------------------------

typedef __attribute__((ext_vector_type(16))) _Float16 v16h;
typedef __attribute__((ext_vector_type(8)))  float    v8f;
typedef _Float16 h16;

#define EE 65536      // edges
#define NN 4096       // nodes
#define GG 64         // graphs
#define MIDD 32       // radial MLP hidden
#define ESTD 0.1f
#define CEU 0.529177211f
#define CUTOFF 4.6f

#define RCHUNKS 19            // 2 (GEMM1) + 16 (GEMM2) + 1 (bias)
#define PACKSZ (RCHUNKS*512)  // halves per packed radial blob

// ----- WMMA fragment index helper (ISA 7.12.2, 16-bit A/B 16x32 / 32x16) ----
// lane L, half-index hi (0..15): K = base(hi) + 8*(L>=16)
__device__ __forceinline__ int wmma_k(int lane, int hi) {
  int p = hi >> 1, pos = hi & 1;
  int base = (p < 4) ? (2 * p + pos) : (16 + 2 * (p - 4) + pos);
  return base + ((lane & 16) ? 8 : 0);
}

__device__ __forceinline__ float bumpf(float x) {
  x = fminf(fmaxf(x, -1.f + 1e-6f), 1.f - 1e-6f);
  return expf(1.f - 1.f / (1.f - x * x));
}
// ordered-int encoding for float atomicMax
__device__ __forceinline__ unsigned fenc(float f) {
  unsigned u = __float_as_uint(f);
  return (u & 0x80000000u) ? ~u : (u | 0x80000000u);
}
__device__ __forceinline__ float fdec(unsigned u) {
  return __uint_as_float((u & 0x80000000u) ? (u & 0x7fffffffu) : ~u);
}

// ---------------------------------------------------------------------------
// Weight pre-pack: emit B-fragments in WMMA lane layout [chunk][lane][hi].
// chunk 0..1 : w1 (16x32, K padded to 32)       -> GEMM1 col-blocks
// chunk 2..17: w2 reshaped (512 x 16), K-chunks -> GEMM2
// chunk 18   : b2 rows as a K<16 chunk          -> bias fold
// ---------------------------------------------------------------------------
__global__ void __launch_bounds__(32)
pack_radial_kernel(const float* __restrict__ w1, const float* __restrict__ w2,
                   const float* __restrict__ b2, h16* __restrict__ out,
                   int co, int ob) {
  const int lane = threadIdx.x;
  const int chunk = blockIdx.x;  // 0..18
  const int n = lane & 15;
  h16 vals[16];
#pragma unroll
  for (int hi = 0; hi < 16; ++hi) {
    int K = wmma_k(lane, hi);
    float v;
    if (chunk < 2) {
      v = (K < 16) ? w1[K * MIDD + chunk * 16 + n] : 0.f;
    } else if (chunk < 18) {
      int g = (chunk - 2) * 32 + K;
      v = w2[(g >> 4) * (co * 16) + (ob * 16 + n) * 16 + (g & 15)];
    } else {
      v = (K < 16) ? b2[(ob * 16 + n) * 16 + K] : 0.f;
    }
    vals[hi] = (h16)v;
  }
#pragma unroll
  for (int hi = 0; hi < 16; ++hi)
    out[(size_t)chunk * 512 + lane * 16 + hi] = vals[hi];
}

// pack mlp_w1 (32x32, full K): 2 col-block fragments
__global__ void __launch_bounds__(32)
pack_mlp_kernel(const float* __restrict__ w1, h16* __restrict__ out) {
  const int lane = threadIdx.x;
  const int cb = blockIdx.x;  // 0..1
  const int n = lane & 15;
#pragma unroll
  for (int hi = 0; hi < 16; ++hi) {
    int K = wmma_k(lane, hi);
    out[(size_t)cb * 512 + lane * 16 + hi] = (h16)w1[K * 32 + cb * 16 + n];
  }
}

// ---------------------------------------------------------------------------
// Fused radial MLP + generated-weight contraction (WMMA hot kernel).
// One wave per 16-edge tile; A-fragments built directly from per-lane loads
// (no LDS except the GEMM1->GEMM2 h transpose); B-fragments are b128 loads
// of the pre-packed blob.
// ---------------------------------------------------------------------------
__global__ void __launch_bounds__(32)
radial_contract_kernel(const h16* __restrict__ e0h, const h16* __restrict__ fbuf,
                       const h16* __restrict__ packed,
                       const float* __restrict__ b1,
                       float* __restrict__ mout, int co, int ob) {
  __shared__ h16 sH[16][32];
  const int lane = threadIdx.x;
  const int tile = blockIdx.x;
  const int row = lane & 15;
  const int grp = lane >> 4;
  const int n = lane & 15;

  // Per-lane direct loads: e0 halves are exactly A-frag elements 0..7
  // (K = grp*8 + j), elements 8..15 are the zero-padded K>=16 region.
  const h16* ep = e0h + (size_t)(tile * 16 + row) * 16 + grp * 8;
  const h16* fp = fbuf + (size_t)(tile * 16 + row) * 16 + grp * 8;
  v16h a = {};
  h16 f8[8];
#pragma unroll
  for (int j = 0; j < 8; ++j) { a[j] = ep[j]; f8[j] = fp[j]; }

  // GEMM1: h = relu(e0 @ w1 + b1)
#pragma unroll
  for (int cb = 0; cb < 2; ++cb) {
    v16h b = *(const v16h*)(packed + (size_t)cb * 512 + lane * 16);
    v8f acc = {};
    acc = __builtin_amdgcn_wmma_f32_16x16x32_f16(false, a, false, b, (short)0,
                                                 acc, false, false);
    float bias = b1[cb * 16 + n];
#pragma unroll
    for (int r = 0; r < 8; ++r) {
      int M = r + ((lane & 16) ? 8 : 0);
      float v = acc[r] + bias;
      sH[M][cb * 16 + n] = (h16)(v > 0.f ? v : 0.f);
    }
  }
  __syncthreads();

  // pull my row's 32 h values into registers (contiguous LDS b128 loads)
  v16h h0, h1;
#pragma unroll
  for (int j = 0; j < 16; ++j) { h0[j] = sH[row][j]; h1[j] = sH[row][16 + j]; }

  // GEMM2: m = (h (x) f) @ w2r, K=512 in 16 chunks; A built as outer product:
  // a[hi] = h[row][2c + (hi>=8)] * f[row][grp*8 + (hi&7)]
  v8f acc = {};
#pragma unroll
  for (int c = 0; c < 16; ++c) {
    h16 hk0 = (c < 8) ? h0[2 * c] : h1[2 * c - 16];
    h16 hk1 = (c < 8) ? h0[2 * c + 1] : h1[2 * c - 15];
    v16h av;
#pragma unroll
    for (int hi = 0; hi < 16; ++hi) av[hi] = (hi < 8 ? hk0 : hk1) * f8[hi & 7];
    v16h b = *(const v16h*)(packed + (size_t)(2 + c) * 512 + lane * 16);
    __builtin_prefetch(packed + (size_t)(3 + c) * 512 + lane * 16, 0, 1);
    acc = __builtin_amdgcn_wmma_f32_16x16x32_f16(false, av, false, b, (short)0,
                                                 acc, false, false);
  }
  // bias chunk: A = [f | 0], B = packed b2 rows
  {
    v16h av = {};
#pragma unroll
    for (int j = 0; j < 8; ++j) av[j] = f8[j];
    v16h b = *(const v16h*)(packed + (size_t)18 * 512 + lane * 16);
    acc = __builtin_amdgcn_wmma_f32_16x16x32_f16(false, av, false, b, (short)0,
                                                 acc, false, false);
  }
#pragma unroll
  for (int r = 0; r < 8; ++r) {
    int M = r + ((lane & 16) ? 8 : 0);
    mout[(size_t)(tile * 16 + M) * co + ob * 16 + n] = acc[r];
  }
}

// ---------------------------------------------------------------------------
// Final readout: m = mA*Y0 + mB; h = silu(m@mlp_w1+b1) (2 WMMA); energy dot.
// A-fragment built directly from two contiguous 8-float runs per lane.
// ---------------------------------------------------------------------------
__global__ void __launch_bounds__(32)
final_mlp_kernel(const float* __restrict__ mA, const float* __restrict__ mB,
                 const float* __restrict__ Y0, const float* __restrict__ coul,
                 const h16* __restrict__ w1pack, const float* __restrict__ b1,
                 const float* __restrict__ w2, const float* __restrict__ b2,
                 const int* __restrict__ eg, float* __restrict__ out) {
  __shared__ float sHf[16][32];
  const int lane = threadIdx.x;
  const int tile = blockIdx.x;
  const int row = lane & 15;
  const int grp = lane >> 4;
  const int n = lane & 15;

  const float y0 = Y0[tile * 16 + row];
  const size_t base = (size_t)(tile * 16 + row) * 32;
  v16h a;
#pragma unroll
  for (int half = 0; half < 2; ++half) {
#pragma unroll
    for (int j = 0; j < 8; ++j) {
      int cidx = half * 16 + grp * 8 + j;  // == wmma_k(lane, half*8+j)
      float v = mA[base + cidx] * y0 + mB[base + cidx];
      a[half * 8 + j] = (h16)v;
    }
  }
#pragma unroll
  for (int cb = 0; cb < 2; ++cb) {
    v16h b = *(const v16h*)(w1pack + (size_t)cb * 512 + lane * 16);
    v8f acc = {};
    acc = __builtin_amdgcn_wmma_f32_16x16x32_f16(false, a, false, b, (short)0,
                                                 acc, false, false);
    float bias = b1[cb * 16 + n];
#pragma unroll
    for (int r = 0; r < 8; ++r) {
      int M = r + ((lane & 16) ? 8 : 0);
      float v = acc[r] + bias;
      sHf[M][cb * 16 + n] = v / (1.f + expf(-v));  // silu
    }
  }
  __syncthreads();

  if (lane < 16) {
    int e = tile * 16 + lane;
    float s = b2[0];
#pragma unroll
    for (int k = 0; k < 32; ++k) s += sHf[lane][k] * w2[k];
    float edge_e = s * Y0[e] + coul[e];
    atomicAdd(&out[eg[e]], edge_e * ESTD);
  }
}

// -------------------------- small VALU kernels ------------------------------
__global__ void geom_kernel(const float* __restrict__ rel,
                            const int* __restrict__ src,
                            const int* __restrict__ dst,
                            const int* __restrict__ species,
                            float* __restrict__ e0f, h16* __restrict__ e0h,
                            float* __restrict__ Y0, float* __restrict__ Y1,
                            float* __restrict__ coul) {
  int e = blockIdx.x * blockDim.x + threadIdx.x;
  if (e >= EE) return;
  float x = rel[e * 3], y = rel[e * 3 + 1], z = rel[e * 3 + 2];
  float d = sqrtf(x * x + y * y + z * z);
  float inv = 1.f / (d + 1e-9f);
  float sc = (d < CUTOFF) ? bumpf(2.f * d / CUTOFF - 1.f) : 0.f;
  float sub = (d < 1.f) ? bumpf(d) : 0.f;  // SCREEN = 1.0
  Y0[e] = sc;
  Y1[e * 3 + 0] = x * inv * sc;
  Y1[e * 3 + 1] = y * inv * sc;
  Y1[e * 3 + 2] = z * inv * sc;
  const float dx = CUTOFF / 14.f;
  e0f[e * 16] = d; e0h[e * 16] = (h16)d;
#pragma unroll
  for (int j = 0; j < 15; ++j) {
    float t = d - dx * (float)j;
    float v = expf(-2.f * t * t / (dx * dx));
    e0f[e * 16 + 1 + j] = v; e0h[e * 16 + 1 + j] = (h16)v;
  }
  float Zs = (float)species[src[e]], Zd = (float)species[dst[e]];
  float cf = Zs * Zd / (2.f * d);
  coul[e] = (sub > 0.f) ? (CEU / ESTD) * cf * sub : 0.f;
}

__global__ void embed_kernel(const int* __restrict__ species,
                             const float* __restrict__ embed,
                             float* __restrict__ f0) {
  int t = blockIdx.x * blockDim.x + threadIdx.x;
  if (t >= NN * 16) return;
  int nI = t >> 4, c = t & 15;
  f0[t] = embed[(species[nI] - 1) * 16 + c];
}

__global__ void gather_f0_kernel(const int* __restrict__ src,
                                 const float* __restrict__ f0,
                                 h16* __restrict__ fb0) {
  int t = blockIdx.x * blockDim.x + threadIdx.x;
  if (t >= EE * 16) return;
  int e = t >> 4, i = t & 15;
  fb0[t] = (h16)f0[src[e] * 16 + i];
}

__global__ void gather_f1_kernel(const int* __restrict__ src,
                                 const float* __restrict__ f1,
                                 const float* __restrict__ Y1,
                                 h16* __restrict__ fbm, h16* __restrict__ fbg) {
  int t = blockIdx.x * blockDim.x + threadIdx.x;
  if (t >= EE * 16) return;
  int e = t >> 4, i = t & 15;
  int s = src[e];
  float a0 = f1[(s * 16 + i) * 3 + 0];
  float a1 = f1[(s * 16 + i) * 3 + 1];
  float a2 = f1[(s * 16 + i) * 3 + 2];
  fbm[0 * EE * 16 + t] = (h16)a0;
  fbm[1 * EE * 16 + t] = (h16)a1;
  fbm[2 * EE * 16 + t] = (h16)a2;
  fbg[t] = (h16)(a0 * Y1[e * 3] + a1 * Y1[e * 3 + 1] + a2 * Y1[e * 3 + 2]);
}

// assemble k0/v0/k1/v1 from radial outputs (+ Y0/Y1 equivariant combination)
__global__ void assemble_kv_kernel(const float* __restrict__ m00,
                                   const float* __restrict__ m01,
                                   const float* __restrict__ m10,
                                   const float* __restrict__ m11s,
                                   const float* __restrict__ m11v,
                                   const float* __restrict__ Y0,
                                   const float* __restrict__ Y1,
                                   float* __restrict__ k0, float* __restrict__ v0,
                                   float* __restrict__ k1, float* __restrict__ v1,
                                   int has1) {
  int t = blockIdx.x * blockDim.x + threadIdx.x;
  if (t >= EE * 16) return;
  int e = t >> 4, o = t & 15;
  float y0 = Y0[e];
  float y1x = Y1[e * 3], y1y = Y1[e * 3 + 1], y1z = Y1[e * 3 + 2];
  float kv0 = m00[t] * y0;
  float s01 = m01[t];
  float kv1x = s01 * y1x, kv1y = s01 * y1y, kv1z = s01 * y1z;
  if (has1) {
    kv0 += m10[t];
    kv1x += m11s[0 * EE * 16 + t] * y0;
    kv1y += m11s[1 * EE * 16 + t] * y0;
    kv1z += m11s[2 * EE * 16 + t] * y0;
    float fvx = m11v[0 * EE * 16 + t];
    float fvy = m11v[1 * EE * 16 + t];
    float fvz = m11v[2 * EE * 16 + t];
    kv1x += y1y * fvz - y1z * fvy;  // cross(Y1, fv)
    kv1y += y1z * fvx - y1x * fvz;
    kv1z += y1x * fvy - y1y * fvx;
  }
  if (o < 8) {
    k0[e * 8 + o] = kv0;
    k1[(e * 8 + o) * 3 + 0] = kv1x;
    k1[(e * 8 + o) * 3 + 1] = kv1y;
    k1[(e * 8 + o) * 3 + 2] = kv1z;
  } else {
    int oo = o - 8;
    v0[e * 8 + oo] = kv0;
    v1[(e * 8 + oo) * 3 + 0] = kv1x;
    v1[(e * 8 + oo) * 3 + 1] = kv1y;
    v1[(e * 8 + oo) * 3 + 2] = kv1z;
  }
}

__global__ void q_kernel(const float* __restrict__ f0, const float* __restrict__ f1,
                         const float* __restrict__ q0w, const float* __restrict__ q1w,
                         float* __restrict__ q0, float* __restrict__ q1, int has1) {
  int t = blockIdx.x * blockDim.x + threadIdx.x;
  if (t >= NN * 8) return;
  int nI = t >> 3, h = t & 7;
  float s = 0.f;
#pragma unroll
  for (int c = 0; c < 16; ++c) s += f0[nI * 16 + c] * q0w[c * 8 + h];
  q0[t] = s;
  if (has1) {
#pragma unroll
    for (int m = 0; m < 3; ++m) {
      float u = 0.f;
      for (int c = 0; c < 16; ++c) u += f1[(nI * 16 + c) * 3 + m] * q1w[c * 8 + h];
      q1[t * 3 + m] = u;
    }
  }
}

__global__ void logits_kernel(const int* __restrict__ dst,
                              const float* __restrict__ q0,
                              const float* __restrict__ q1,
                              const float* __restrict__ k0,
                              const float* __restrict__ k1,
                              float* __restrict__ lg, float invs, int has1) {
  int t = blockIdx.x * blockDim.x + threadIdx.x;
  if (t >= EE * 8) return;
  int e = t >> 3, h = t & 7;
  int d = dst[e];
  float v = q0[d * 8 + h] * k0[t];
  if (has1) {
#pragma unroll
    for (int m = 0; m < 3; ++m) v += q1[(d * 8 + h) * 3 + m] * k1[t * 3 + m];
  }
  lg[t] = v * invs;
}

__global__ void segmax_kernel(const int* __restrict__ dst,
                              const float* __restrict__ lg,
                              unsigned* __restrict__ mmax) {
  int t = blockIdx.x * blockDim.x + threadIdx.x;
  if (t >= EE * 8) return;
  int e = t >> 3, h = t & 7;
  atomicMax(&mmax[dst[e] * 8 + h], fenc(lg[t]));
}

__global__ void expsum_kernel(const int* __restrict__ dst,
                              const float* __restrict__ Y0,
                              const unsigned* __restrict__ mmax,
                              float* __restrict__ lg, float* __restrict__ den) {
  int t = blockIdx.x * blockDim.x + threadIdx.x;
  if (t >= EE * 8) return;
  int e = t >> 3, h = t & 7;
  float m = fdec(mmax[dst[e] * 8 + h]);
  if (!(m >= -3.0e38f && m <= 3.0e38f)) m = 0.f;  // where(isfinite(m), m, 0)
  float w = Y0[e] * expf(lg[t] - m);
  lg[t] = w;
  atomicAdd(&den[dst[e] * 8 + h], w);
}

__global__ void aggregate_kernel(const int* __restrict__ dst,
                                 const float* __restrict__ lg,
                                 const float* __restrict__ den,
                                 const float* __restrict__ v0,
                                 const float* __restrict__ v1,
                                 float* __restrict__ o0, float* __restrict__ o1) {
  int t = blockIdx.x * blockDim.x + threadIdx.x;
  if (t >= EE * 8) return;
  int e = t >> 3, h = t & 7;
  int d = dst[e];
  float alpha = lg[t] / (den[d * 8 + h] + 1e-9f);
  atomicAdd(&o0[d * 8 + h], alpha * v0[t]);
#pragma unroll
  for (int m = 0; m < 3; ++m)
    atomicAdd(&o1[(d * 8 + h) * 3 + m], alpha * v1[t * 3 + m]);
}

__global__ void node_update_kernel(const float* __restrict__ o0,
                                   const float* __restrict__ o1,
                                   const float* __restrict__ f0,
                                   const float* __restrict__ f1,
                                   const float* __restrict__ p0b,
                                   const float* __restrict__ p0w,
                                   const float* __restrict__ p1w,
                                   float* __restrict__ f0n, float* __restrict__ f1n,
                                   int has1) {
  int t = blockIdx.x * blockDim.x + threadIdx.x;
  if (t >= NN * 16) return;
  int nI = t >> 4, o = t & 15;
  float s = p0b[o];
#pragma unroll
  for (int c = 0; c < 8; ++c) s += o0[nI * 8 + c] * p0w[c * 16 + o];
#pragma unroll
  for (int c = 0; c < 16; ++c) s += f0[nI * 16 + c] * p0w[(8 + c) * 16 + o];
  f0n[t] = s;
#pragma unroll
  for (int m = 0; m < 3; ++m) {
    float u = 0.f;
    for (int c = 0; c < 8; ++c) u += o1[(nI * 8 + c) * 3 + m] * p1w[c * 16 + o];
    if (has1)
      for (int c = 0; c < 16; ++c)
        u += f1[(nI * 16 + c) * 3 + m] * p1w[(8 + c) * 16 + o];
    f1n[t * 3 + m] = u;
  }
}

__global__ void e0_update_kernel(const int* __restrict__ src,
                                 const float* __restrict__ f0,
                                 float* __restrict__ e0f, h16* __restrict__ e0h) {
  int t = blockIdx.x * blockDim.x + threadIdx.x;
  if (t >= EE * 16) return;
  int e = t >> 4, i = t & 15;
  float v = e0f[t] + f0[src[e] * 16 + i];
  e0f[t] = v;
  e0h[t] = (h16)v;
}

__global__ void fill_f32_kernel(float* p, int n, float v) {
  int t = blockIdx.x * blockDim.x + threadIdx.x;
  if (t < n) p[t] = v;
}
__global__ void fill_u32_kernel(unsigned* p, int n, unsigned v) {
  int t = blockIdx.x * blockDim.x + threadIdx.x;
  if (t < n) p[t] = v;
}

// ---------------------------------------------------------------------------
// Host orchestration.
// Parameter flattening assumption (documented): top-level setup_inputs() in
// insertion order (src,dst,edge_graph,species,rel_pos,params); params pytree
// flattened with sorted dict keys, lists in order:
//   5 embed | 6..13 final_radial{00,10}x{b1,b2,w1,w2}
//   14..25 layer0: p0b,p0w,p1w,q0, r00{b1,b2,w1,w2}, r01{...}
//   layer li>=1 at 26+(li-1)*25: p0b,p0w,p1w,q0,q1, r00,r01,r10,r11s,r11v
//   101 mlp_b1 | 102 mlp_b2 | 103 mlp_w1 | 104 mlp_w2
// ---------------------------------------------------------------------------
extern "C" void kernel_launch(void* const* d_in, const int* in_sizes, int n_in,
                              void* d_out, int out_size, void* d_ws, size_t ws_size,
                              hipStream_t stream) {
  (void)in_sizes; (void)n_in; (void)out_size; (void)ws_size;
  const int* src = (const int*)d_in[0];
  const int* dst = (const int*)d_in[1];
  const int* eg = (const int*)d_in[2];
  const int* species = (const int*)d_in[3];
  const float* rel = (const float*)d_in[4];
  auto P = [&](int i) { return (const float*)d_in[i]; };
  float* out = (float*)d_out;

  // workspace carve-out (256B aligned)
  char* w = (char*)d_ws;
  size_t off = 0;
  auto alloc = [&](size_t bytes) -> void* {
    void* p = w + off;
    off = (off + bytes + 255) & ~(size_t)255;
    return p;
  };
  float* e0f = (float*)alloc((size_t)EE * 16 * 4);
  h16* e0h = (h16*)alloc((size_t)EE * 16 * 2);
  float* Y0 = (float*)alloc((size_t)EE * 4);
  float* Y1 = (float*)alloc((size_t)EE * 12);
  float* coul = (float*)alloc((size_t)EE * 4);
  h16* fb0 = (h16*)alloc((size_t)EE * 16 * 2);
  h16* fbg = (h16*)alloc((size_t)EE * 16 * 2);
  h16* fbm = (h16*)alloc((size_t)EE * 16 * 2 * 3);
  float* mpool = (float*)alloc((size_t)EE * 16 * 4 * 9);
  float* m00 = mpool;
  float* m01 = mpool + (size_t)EE * 16;
  float* m10 = mpool + (size_t)EE * 16 * 2;
  float* m11s = mpool + (size_t)EE * 16 * 3;  // 3 slices
  float* m11v = mpool + (size_t)EE * 16 * 6;  // 3 slices
  float* mA = mpool;                          // final reuse (E x 32)
  float* mB = mpool + (size_t)EE * 32;
  float* k0 = (float*)alloc((size_t)EE * 8 * 4);
  float* v0 = (float*)alloc((size_t)EE * 8 * 4);
  float* k1 = (float*)alloc((size_t)EE * 24 * 4);
  float* v1 = (float*)alloc((size_t)EE * 24 * 4);
  float* lg = (float*)alloc((size_t)EE * 8 * 4);
  float* den = (float*)alloc((size_t)NN * 8 * 4);
  unsigned* mmax = (unsigned*)alloc((size_t)NN * 8 * 4);
  float* o0 = (float*)alloc((size_t)NN * 8 * 4);
  float* o1 = (float*)alloc((size_t)NN * 24 * 4);
  float* f0A = (float*)alloc((size_t)NN * 16 * 4);
  float* f0B = (float*)alloc((size_t)NN * 16 * 4);
  float* f1A = (float*)alloc((size_t)NN * 48 * 4);
  float* f1B = (float*)alloc((size_t)NN * 48 * 4);
  float* q0b = (float*)alloc((size_t)NN * 8 * 4);
  float* q1b = (float*)alloc((size_t)NN * 24 * 4);
  h16* packs = (h16*)alloc((size_t)21 * PACKSZ * 2);  // 21 radial blobs
  h16* mlppack = (h16*)alloc((size_t)2 * 512 * 2);

  auto G = [](int n) { return dim3((unsigned)((n + 255) / 256)); };
  const dim3 B256(256);
  const dim3 WTILE(32);
  const dim3 GTILE((unsigned)(EE / 16));
  const dim3 GPACK((unsigned)RCHUNKS);

  auto pack_radial = [&](int pb, int co, int ob, int slot) {
    pack_radial_kernel<<<GPACK, WTILE, 0, stream>>>(
        P(pb + 2), P(pb + 3), P(pb + 1), packs + (size_t)slot * PACKSZ, co, ob);
  };
  auto launch_radial = [&](int pb, int slot, const h16* f, float* mout, int co,
                           int ob) {
    radial_contract_kernel<<<GTILE, WTILE, 0, stream>>>(
        e0h, f, packs + (size_t)slot * PACKSZ, P(pb), mout, co, ob);
  };

  // ----- pre-pack all weight fragments (slots) -----
  // slot 0,1: layer0 r00,r01 | slot 2+(li-1)*5 ..: r00,r01,r10,r11s,r11v
  // slot 17..20: final fr00 ob0/ob1, fr10 ob0/ob1
  pack_radial(18, 16, 0, 0);
  pack_radial(22, 16, 0, 1);
  for (int li = 1; li < 4; ++li) {
    int L = 26 + (li - 1) * 25;
    int s = 2 + (li - 1) * 5;
    pack_radial(L + 5, 16, 0, s + 0);
    pack_radial(L + 9, 16, 0, s + 1);
    pack_radial(L + 13, 16, 0, s + 2);
    pack_radial(L + 17, 16, 0, s + 3);
    pack_radial(L + 21, 16, 0, s + 4);
  }
  pack_radial(6, 32, 0, 17);
  pack_radial(6, 32, 1, 18);
  pack_radial(10, 32, 0, 19);
  pack_radial(10, 32, 1, 20);
  pack_mlp_kernel<<<dim3(2), WTILE, 0, stream>>>(P(103), mlppack);

  geom_kernel<<<G(EE), B256, 0, stream>>>(rel, src, dst, species, e0f, e0h, Y0,
                                          Y1, coul);
  embed_kernel<<<G(NN * 16), B256, 0, stream>>>(species, P(5), f0A);

  float* f0c = f0A; float* f0n = f0B;
  float* f1c = f1A; float* f1n = f1B;

  for (int li = 0; li < 4; ++li) {
    const int has1 = (li > 0);
    const int L = has1 ? (26 + (li - 1) * 25) : 0;
    const int r00 = has1 ? (L + 5) : 18;
    const int r01 = has1 ? (L + 9) : 22;
    const int q0i = has1 ? (L + 3) : 17;
    const int p0bI = has1 ? L : 14, p0wI = has1 ? (L + 1) : 15,
              p1wI = has1 ? (L + 2) : 16;
    const int s = has1 ? (2 + (li - 1) * 5) : 0;

    gather_f0_kernel<<<G(EE * 16), B256, 0, stream>>>(src, f0c, fb0);
    if (has1)
      gather_f1_kernel<<<G(EE * 16), B256, 0, stream>>>(src, f1c, Y1, fbm, fbg);

    launch_radial(r00, has1 ? s + 0 : 0, fb0, m00, 16, 0);
    launch_radial(r01, has1 ? s + 1 : 1, fb0, m01, 16, 0);
    if (has1) {
      launch_radial(L + 13, s + 2, fbg, m10, 16, 0);
      for (int m = 0; m < 3; ++m)
        launch_radial(L + 17, s + 3, fbm + (size_t)m * EE * 16,
                      m11s + (size_t)m * EE * 16, 16, 0);
      for (int m = 0; m < 3; ++m)
        launch_radial(L + 21, s + 4, fbm + (size_t)m * EE * 16,
                      m11v + (size_t)m * EE * 16, 16, 0);
    }

    assemble_kv_kernel<<<G(EE * 16), B256, 0, stream>>>(
        m00, m01, m10, m11s, m11v, Y0, Y1, k0, v0, k1, v1, has1);

    q_kernel<<<G(NN * 8), B256, 0, stream>>>(f0c, f1c, P(q0i),
                                             has1 ? P(L + 4) : P(q0i), q0b, q1b,
                                             has1);
    logits_kernel<<<G(EE * 8), B256, 0, stream>>>(dst, q0b, q1b, k0, k1, lg,
                                                  has1 ? 0.5f : 1.0f, has1);

    fill_u32_kernel<<<G(NN * 8), B256, 0, stream>>>(mmax, NN * 8, 0x007FFFFFu);
    fill_f32_kernel<<<G(NN * 8), B256, 0, stream>>>(den, NN * 8, 0.f);
    fill_f32_kernel<<<G(NN * 8), B256, 0, stream>>>(o0, NN * 8, 0.f);
    fill_f32_kernel<<<G(NN * 24), B256, 0, stream>>>(o1, NN * 24, 0.f);

    segmax_kernel<<<G(EE * 8), B256, 0, stream>>>(dst, lg, mmax);
    expsum_kernel<<<G(EE * 8), B256, 0, stream>>>(dst, Y0, mmax, lg, den);
    aggregate_kernel<<<G(EE * 8), B256, 0, stream>>>(dst, lg, den, v0, v1, o0, o1);

    node_update_kernel<<<G(NN * 16), B256, 0, stream>>>(
        o0, o1, f0c, f1c, P(p0bI), P(p0wI), P(p1wI), f0n, f1n, has1);
    e0_update_kernel<<<G(EE * 16), B256, 0, stream>>>(src, f0n, e0f, e0h);

    float* t0 = f0c; f0c = f0n; f0n = t0;
    float* t1 = f1c; f1c = f1n; f1n = t1;
  }

  // ----- final readout -----
  gather_f0_kernel<<<G(EE * 16), B256, 0, stream>>>(src, f0c, fb0);
  gather_f1_kernel<<<G(EE * 16), B256, 0, stream>>>(src, f1c, Y1, fbm, fbg);
  launch_radial(6, 17, fb0, mA, 32, 0);
  launch_radial(6, 18, fb0, mA, 32, 1);
  launch_radial(10, 19, fbg, mB, 32, 0);
  launch_radial(10, 20, fbg, mB, 32, 1);
  fill_f32_kernel<<<G(GG), B256, 0, stream>>>(out, GG, 0.f);
  final_mlp_kernel<<<GTILE, WTILE, 0, stream>>>(mA, mB, Y0, coul, mlppack,
                                                P(101), P(104), P(102), eg, out);
}
